// QuantumGraphConv_40931038331194
// MI455X (gfx1250) — matrix operand order repaired and checked by reading, not compiled, not run on hardware
//
#include <hip/hip_runtime.h>
#include <hip/hip_bf16.h>

typedef __attribute__((ext_vector_type(16))) _Float16 v16h;
typedef __attribute__((ext_vector_type(8)))  _Float16 v8h;
typedef __attribute__((ext_vector_type(8)))  float    v8f;

#define N_NODES 100000
#define N_EDGES 500000

#define WMMA_F16(a,b,c) __builtin_amdgcn_wmma_f32_16x16x32_f16(false,(a),false,(b),(short)0,(c),false,false)

__device__ __forceinline__ v8f splat8(float x) {
    v8f r;
#pragma unroll
    for (int i = 0; i < 8; ++i) r[i] = x;
    return r;
}

// B fragment (32x16, K x N) for 16x16x32 f16 WMMA, from row-major W (outFeat x inFeat),
// i.e. B[k][n] = W[n0+n][k0+k]. Lane L: column n = L%16; lanes<16 hold K=k0+0..7 and
// k0+16..23, lanes>=16 hold +8 -> two 16B reads per lane.
__device__ __forceinline__ v16h load_b_frag(const _Float16* __restrict__ W, int ldk,
                                            int n0, int k0, int lane) {
    int n = n0 + (lane & 15);
    int k = k0 + ((lane & 16) ? 8 : 0);
    const _Float16* p = W + (size_t)n * ldk + k;
    v8h lo = *(const v8h*)p;
    v8h hi = *(const v8h*)(p + 16);
    return __builtin_shufflevector(lo, hi, 0,1,2,3,4,5,6,7,8,9,10,11,12,13,14,15);
}

// A fragment (16x32, M x K) from LDS row-major tile S[16][ld].
// Lane L: row m = L%16; lanes<16 hold K = k0+0..7 and k0+16..23, lanes>=16 hold +8.
__device__ __forceinline__ v16h load_a_frag(const _Float16* S, int ld, int k0, int lane) {
    int m = lane & 15;
    int off = (lane & 16) ? 8 : 0;
    const _Float16* p = S + m * ld + k0 + off;
    v8h lo = *(const v8h*)p;
    v8h hi = *(const v8h*)(p + 16);
    return __builtin_shufflevector(lo, hi, 0,1,2,3,4,5,6,7,8,9,10,11,12,13,14,15);
}

__device__ __forceinline__ float red16(float v) {
    v += __shfl_xor(v, 1, 16);
    v += __shfl_xor(v, 2, 16);
    v += __shfl_xor(v, 4, 16);
    v += __shfl_xor(v, 8, 16);
    return v;
}

// ---------------------------------------------------------------------------
// Weight f32 -> f16 conversion
// ---------------------------------------------------------------------------
__global__ void cvt_kernel(const float* __restrict__ src, _Float16* __restrict__ dst, int n) {
    int i = blockIdx.x * 256 + threadIdx.x;
    if (i < n) dst[i] = (_Float16)src[i];
}

// ---------------------------------------------------------------------------
// Node phase: qs = tanh(LN(x@sp_w.T+sp_b)); amp/ph linears; qf = [amp*cos, amp*sin]
// wave-per-node; weights cached in LDS as f16 with +2 half padding (bank-conflict free).
// Also zeroes agg (segment-sum accumulator).
// ---------------------------------------------------------------------------
#define NODE_WAVES 8
__global__ void __launch_bounds__(256) node_kernel(
    const float* __restrict__ x,
    const float* __restrict__ spw, const float* __restrict__ spb,
    const float* __restrict__ ln1g, const float* __restrict__ ln1b,
    const float* __restrict__ pgates,
    const float* __restrict__ ampw, const float* __restrict__ ampb,
    const float* __restrict__ phw,  const float* __restrict__ phb,
    _Float16* __restrict__ qf, float* __restrict__ agg)
{
    __shared__ _Float16 sSp[128][130];
    __shared__ _Float16 sAmp[64][66];
    __shared__ _Float16 sPh[64][66];
    for (int i = threadIdx.x; i < 128 * 128; i += blockDim.x)
        sSp[i >> 7][i & 127] = (_Float16)spw[i];
    for (int i = threadIdx.x; i < 64 * 64; i += blockDim.x) {
        sAmp[i >> 6][i & 63] = (_Float16)ampw[i];
        sPh[i >> 6][i & 63]  = (_Float16)phw[i];
    }
    __syncthreads();

    const int lane = threadIdx.x & 31;
    const int wv   = threadIdx.x >> 5;

    float pg0 = 0.f, pg1 = 0.f;
#pragma unroll
    for (int q = 0; q < 8; ++q) {
        pg0 += pgates[q * 64 + lane];
        pg1 += pgates[q * 64 + 32 + lane];
    }

    for (int node = blockIdx.x * NODE_WAVES + wv; node < N_NODES;
         node += gridDim.x * NODE_WAVES) {
        float xr[4], acc[4];
#pragma unroll
        for (int c = 0; c < 4; ++c) {
            xr[c]  = x[(size_t)node * 128 + c * 32 + lane];
            acc[c] = spb[c * 32 + lane];
            agg[(size_t)node * 128 + c * 32 + lane] = 0.f;
        }
        // y[j] = x @ sp_w[j], j = lane + 32c, via shfl broadcast of x
#pragma unroll
        for (int c2 = 0; c2 < 4; ++c2) {
            for (int kk = 0; kk < 32; ++kk) {
                float xk = __shfl(xr[c2], kk, 32);
                int k = c2 * 32 + kk;
#pragma unroll
                for (int c = 0; c < 4; ++c)
                    acc[c] += xk * (float)sSp[c * 32 + lane][k];
            }
        }
        // layernorm over 128
        float s  = acc[0] + acc[1] + acc[2] + acc[3];
        float s2 = acc[0]*acc[0] + acc[1]*acc[1] + acc[2]*acc[2] + acc[3]*acc[3];
#pragma unroll
        for (int off = 1; off < 32; off <<= 1) {
            s  += __shfl_xor(s,  off, 32);
            s2 += __shfl_xor(s2, off, 32);
        }
        float mean = s * (1.f / 128.f);
        float var  = s2 * (1.f / 128.f) - mean * mean;
        float rstd = rsqrtf(var + 1e-5f);
        float qs[4];
#pragma unroll
        for (int c = 0; c < 4; ++c) {
            int j = c * 32 + lane;
            qs[c] = tanhf((acc[c] - mean) * rstd * ln1g[j] + ln1b[j]);
        }
        float r0 = qs[0], r1 = qs[1], i0 = qs[2], i1 = qs[3];
        float a0 = ampb[lane], a1 = ampb[32 + lane];
        float p0 = phb[lane],  p1 = phb[32 + lane];
#pragma unroll
        for (int half = 0; half < 2; ++half) {
            float rsrc = half ? r1 : r0;
            float isrc = half ? i1 : i0;
            for (int kk = 0; kk < 32; ++kk) {
                float ri = __shfl(rsrc, kk, 32);
                float ii = __shfl(isrc, kk, 32);
                int i = half * 32 + kk;
                a0 += ri * (float)sAmp[lane][i];
                a1 += ri * (float)sAmp[32 + lane][i];
                p0 += ii * (float)sPh[lane][i];
                p1 += ii * (float)sPh[32 + lane][i];
            }
        }
        float amp0 = 1.f / (1.f + __expf(-a0));
        float amp1 = 1.f / (1.f + __expf(-a1));
        float ang0 = tanhf(p0) * 3.14159265358979f + r0 * pg0;
        float ang1 = tanhf(p1) * 3.14159265358979f + r1 * pg1;
        _Float16* qrow = qf + (size_t)node * 128;
        qrow[lane]      = (_Float16)(amp0 * __cosf(ang0));
        qrow[32 + lane] = (_Float16)(amp1 * __cosf(ang1));
        qrow[64 + lane] = (_Float16)(amp0 * __sinf(ang0));
        qrow[96 + lane] = (_Float16)(amp1 * __sinf(ang1));
    }
}

// ---------------------------------------------------------------------------
// Edge phase: wave-per-16-edge tile, all GEMMs on v_wmma_f32_16x16x32_f16.
// ---------------------------------------------------------------------------
#define EW  2      // waves per block
#define LDA 264    // [x_i|x_j] tile row stride (256 + 8 pad halves)
#define LDE 264    // ent tile row stride
#define LDM 136    // msum tile row stride (128 + 8)

__global__ void __launch_bounds__(EW * 32) edge_kernel(
    const _Float16* __restrict__ qf,
    const int* __restrict__ eidx,
    const _Float16* __restrict__ entw, const float* __restrict__ entb,
    const _Float16* __restrict__ qw,   const float* __restrict__ qb,
    const _Float16* __restrict__ kw,   const float* __restrict__ kb,
    const _Float16* __restrict__ vw,   const float* __restrict__ vb,
    const _Float16* __restrict__ measw,const float* __restrict__ measb,
    const _Float16* __restrict__ outw, const float* __restrict__ outb,
    float* __restrict__ agg)
{
    __shared__ _Float16 sA[EW][16][LDA];
    __shared__ _Float16 sE[EW][16][LDE];
    __shared__ _Float16 sM[EW][16][LDM];

    const int lane = threadIdx.x & 31;
    const int wv   = threadIdx.x >> 5;
    const int e0   = (blockIdx.x * EW + wv) * 16;
    if (e0 >= N_EDGES) return;
    const int valid = min(16, N_EDGES - e0);
    const int ln    = lane & 15;
    const int mbase = (lane & 16) ? 8 : 0;

    // ---- gather x_cat = [qf[tgt] | qf[src]] into sA (f16, 16 x 256) ----
    {
        int em   = lane & 15;          // edge within tile
        int part = lane >> 4;          // 0 -> x_i (tgt), 1 -> x_j (src)
        int e    = e0 + min(em, valid - 1);
        int node = (part == 0) ? eidx[N_EDGES + e] : eidx[e];
        const float4* src4 = (const float4*)(qf + (size_t)node * 128);
        float4* dst4 = (float4*)(&sA[wv][em][part * 128]);
        bool ok = em < valid;
        float4 z = make_float4(0.f, 0.f, 0.f, 0.f);
#pragma unroll
        for (int j = 0; j < 16; ++j) dst4[j] = ok ? src4[j] : z;
    }
    // in-wave LDS ordering: DS ops are in-order per wave (no cross-wave sharing).

    // ---- ent = x_cat @ ent_w.T + ent_b -> sE (f16) : 16 tiles x 8 ksteps ----
#pragma unroll 1
    for (int t = 0; t < 16; ++t) {
        int n0 = t * 16;
        v8f acc = splat8(entb[n0 + ln]);
#pragma unroll
        for (int k0 = 0; k0 < 256; k0 += 32) {
            v16h a = load_a_frag(&sA[wv][0][0], LDA, k0, lane);
            acc = WMMA_F16(a, load_b_frag(entw, 256, n0, k0, lane), acc);
        }
        int n = n0 + ln;
#pragma unroll
        for (int r = 0; r < 8; ++r)
            sE[wv][mbase + r][n] = (_Float16)acc[r];
    }

    // ---- q,k,v GEMMs + per-row cross-head softmax -> vreg = qm (C layout) ----
    v8f vreg[8];
    v8f hacc[4];
#pragma unroll
    for (int h = 0; h < 4; ++h) hacc[h] = splat8(0.f);

#pragma unroll
    for (int t = 0; t < 8; ++t) {
        int n0 = t * 16;
        v8f qa = splat8(qb[n0 + ln]);
        v8f ka = splat8(kb[n0 + ln]);
        v8f va = splat8(vb[n0 + ln]);
#pragma unroll
        for (int k0 = 0; k0 < 128; k0 += 32) {
            v16h a = load_a_frag(&sA[wv][0][0], LDA, k0, lane);   // x_i columns
            qa = WMMA_F16(a, load_b_frag(qw, 128, n0, k0, lane), qa);
            ka = WMMA_F16(a, load_b_frag(kw, 128, n0, k0, lane), ka);
            va = WMMA_F16(a, load_b_frag(vw, 128, n0, k0, lane), va);
        }
        vreg[t] = va;
#pragma unroll
        for (int r = 0; r < 8; ++r) hacc[t >> 1][r] += qa[r] * ka[r];
    }
    const float sc = 0.17677669529663687f;  // 1/sqrt(32)
#pragma unroll
    for (int r = 0; r < 8; ++r) {
        float s0 = red16(hacc[0][r]) * sc;
        float s1 = red16(hacc[1][r]) * sc;
        float s2 = red16(hacc[2][r]) * sc;
        float s3 = red16(hacc[3][r]) * sc;
        float mx = fmaxf(fmaxf(s0, s1), fmaxf(s2, s3));
        float ex0 = __expf(s0 - mx), ex1 = __expf(s1 - mx);
        float ex2 = __expf(s2 - mx), ex3 = __expf(s3 - mx);
        float inv = 1.f / (ex0 + ex1 + ex2 + ex3);
        float w4[4] = {ex0 * inv, ex1 * inv, ex2 * inv, ex3 * inv};
#pragma unroll
        for (int t = 0; t < 8; ++t) vreg[t][r] *= w4[t >> 1];
    }

    // ---- measured = mean_k (ei@meas_w[k].T + b) * (ej@meas_w[k].T + b) ----
#pragma unroll
    for (int t = 0; t < 8; ++t) {
        int n0 = t * 16;
#pragma unroll 1
        for (int k = 0; k < 4; ++k) {
            float bm = measb[k * 128 + n0 + ln];
            v8f mi = splat8(bm), mj = splat8(bm);
#pragma unroll
            for (int k0 = 0; k0 < 128; k0 += 32) {
                v16h b  = load_b_frag(measw + k * 16384, 128, n0, k0, lane);
                v16h ai = load_a_frag(&sE[wv][0][0],   LDE, k0, lane);  // ei
                mi = WMMA_F16(ai, b, mi);
                v16h aj = load_a_frag(&sE[wv][0][128], LDE, k0, lane);  // ej
                mj = WMMA_F16(aj, b, mj);
            }
#pragma unroll
            for (int r = 0; r < 8; ++r) vreg[t][r] += 0.25f * mi[r] * mj[r];
        }
    }

    // ---- msum -> sM (f16) ----
#pragma unroll
    for (int t = 0; t < 8; ++t)
#pragma unroll
        for (int r = 0; r < 8; ++r)
            sM[wv][mbase + r][t * 16 + ln] = (_Float16)vreg[t][r];

    // ---- msg = msum @ out_w.T + out_b ; segment-sum via f32 atomics ----
#pragma unroll 1
    for (int t = 0; t < 8; ++t) {
        int n0 = t * 16;
        v8f acc = splat8(outb[n0 + ln]);
#pragma unroll
        for (int k0 = 0; k0 < 128; k0 += 32) {
            v16h a = load_a_frag(&sM[wv][0][0], LDM, k0, lane);
            acc = WMMA_F16(a, load_b_frag(outw, 128, n0, k0, lane), acc);
        }
#pragma unroll
        for (int r = 0; r < 8; ++r) {
            int m = mbase + r;
            if (m < valid) {
                int tgt = eidx[N_EDGES + e0 + m];
                atomicAdd(&agg[(size_t)tgt * 128 + n0 + ln], acc[r]);
            }
        }
    }
}

// ---------------------------------------------------------------------------
// Final: gelu(layernorm(agg)) in-place on d_out (wave-per-node)
// ---------------------------------------------------------------------------
__global__ void __launch_bounds__(256) final_kernel(
    float* __restrict__ agg, const float* __restrict__ g,
    const float* __restrict__ b, float* __restrict__ out)
{
    const int lane = threadIdx.x & 31;
    const int wv   = threadIdx.x >> 5;
    int node = blockIdx.x * 8 + wv;
    if (node >= N_NODES) return;
    float v[4];
#pragma unroll
    for (int c = 0; c < 4; ++c) v[c] = agg[(size_t)node * 128 + c * 32 + lane];
    float s  = v[0] + v[1] + v[2] + v[3];
    float s2 = v[0]*v[0] + v[1]*v[1] + v[2]*v[2] + v[3]*v[3];
#pragma unroll
    for (int off = 1; off < 32; off <<= 1) {
        s  += __shfl_xor(s,  off, 32);
        s2 += __shfl_xor(s2, off, 32);
    }
    float mean = s * (1.f / 128.f);
    float var  = s2 * (1.f / 128.f) - mean * mean;
    float rstd = rsqrtf(var + 1e-5f);
#pragma unroll
    for (int c = 0; c < 4; ++c) {
        int j = c * 32 + lane;
        float y = (v[c] - mean) * rstd * g[j] + b[j];
        out[(size_t)node * 128 + j] = 0.5f * y * (1.f + erff(y * 0.70710678118654752f));
    }
}

// ---------------------------------------------------------------------------
extern "C" void kernel_launch(void* const* d_in, const int* in_sizes, int n_in,
                              void* d_out, int out_size, void* d_ws, size_t ws_size,
                              hipStream_t stream) {
    const float* x      = (const float*)d_in[0];
    const int*   eidx   = (const int*)d_in[1];
    const float* spw    = (const float*)d_in[2];
    const float* spb    = (const float*)d_in[3];
    const float* ln1g   = (const float*)d_in[4];
    const float* ln1b   = (const float*)d_in[5];
    const float* pgates = (const float*)d_in[6];
    const float* ampw   = (const float*)d_in[7];
    const float* ampb   = (const float*)d_in[8];
    const float* phw    = (const float*)d_in[9];
    const float* phb    = (const float*)d_in[10];
    const float* entw   = (const float*)d_in[11];
    const float* entb   = (const float*)d_in[12];
    const float* qwf    = (const float*)d_in[13];
    const float* qb     = (const float*)d_in[14];
    const float* kwf    = (const float*)d_in[15];
    const float* kb     = (const float*)d_in[16];
    const float* vwf    = (const float*)d_in[17];
    const float* vb     = (const float*)d_in[18];
    const float* measwf = (const float*)d_in[19];
    const float* measb  = (const float*)d_in[20];
    const float* outwf  = (const float*)d_in[21];
    const float* outb   = (const float*)d_in[22];
    const float* ln2g   = (const float*)d_in[23];
    const float* ln2b   = (const float*)d_in[24];

    char* ws = (char*)d_ws;
    _Float16* qf = (_Float16*)ws;                         // N*128 f16 = 25.6 MB
    _Float16* wh = (_Float16*)(ws + (size_t)N_NODES * 128 * 2);
    _Float16* entw_h  = wh;                 // 65536
    _Float16* qw_h    = entw_h + 65536;     // 16384
    _Float16* kw_h    = qw_h + 16384;
    _Float16* vw_h    = kw_h + 16384;
    _Float16* measw_h = vw_h + 16384;       // 65536
    _Float16* outw_h  = measw_h + 65536;    // 16384
    float* agg = (float*)d_out;             // accumulate segment-sum in d_out

    cvt_kernel<<<(65536 + 255) / 256, 256, 0, stream>>>(entw,   entw_h,  65536);
    cvt_kernel<<<(16384 + 255) / 256, 256, 0, stream>>>(qwf,    qw_h,    16384);
    cvt_kernel<<<(16384 + 255) / 256, 256, 0, stream>>>(kwf,    kw_h,    16384);
    cvt_kernel<<<(16384 + 255) / 256, 256, 0, stream>>>(vwf,    vw_h,    16384);
    cvt_kernel<<<(65536 + 255) / 256, 256, 0, stream>>>(measwf, measw_h, 65536);
    cvt_kernel<<<(16384 + 255) / 256, 256, 0, stream>>>(outwf,  outw_h,  16384);

    node_kernel<<<1024, 256, 0, stream>>>(x, spw, spb, ln1g, ln1b, pgates,
                                          ampw, ampb, phw, phb, qf, agg);

    edge_kernel<<<(N_EDGES + EW * 16 - 1) / (EW * 16), EW * 32, 0, stream>>>(
        qf, eidx, entw_h, entb, qw_h, qb, kw_h, kb, vw_h, vb,
        measw_h, measb, outw_h, outb, agg);

    final_kernel<<<(N_NODES + 7) / 8, 256, 0, stream>>>(agg, ln2g, ln2b, (float*)d_out);
}
